// MaxKSAGEConv_35381940584975
// MI455X (gfx1250) — matrix operand order repaired and checked by reading, not compiled, not run on hardware
//
#include <hip/hip_runtime.h>
#include <hip/hip_bf16.h>
#include <stdint.h>

typedef __attribute__((ext_vector_type(16))) __bf16 v16bf;
typedef __attribute__((ext_vector_type(8)))  float  v8f;

#define D   256   // D_IN == D_OUT
#define DK  512   // concatenated K: [self | neigh]

// ---------------------------------------------------------------- zero ws ---
__global__ void sage_zero4_kernel(float4* __restrict__ p, long n4) {
    long i = (long)blockIdx.x * blockDim.x + threadIdx.x;
    if (i < n4) p[i] = make_float4(0.f, 0.f, 0.f, 0.f);
}
__global__ void sage_zero_kernel(float* __restrict__ p, long n) {
    long i = (long)blockIdx.x * blockDim.x + threadIdx.x;
    if (i < n) p[i] = 0.0f;
}

// ------------------------------------------- pack [W_self | W_neigh] bf16 ---
__global__ void sage_pack_w_kernel(const float* __restrict__ w_neigh,
                                   const float* __restrict__ w_self,
                                   __bf16* __restrict__ wcat) {
    int idx = blockIdx.x * blockDim.x + threadIdx.x;   // 0 .. 256*512-1
    int n = idx >> 9;
    int k = idx & (DK - 1);
    float v = (k < D) ? w_self[n * D + k] : w_neigh[n * D + (k - D)];
    wcat[idx] = (__bf16)v;
}

// ------------------------------------- edge scatter: one wave32 per edge ---
__global__ void sage_scatter_kernel(const float* __restrict__ feat,
                                    const int*  __restrict__ src,
                                    const int*  __restrict__ dst,
                                    float* __restrict__ neigh,
                                    float* __restrict__ deg, int nedges) {
    int e    = (blockIdx.x * blockDim.x + threadIdx.x) >> 5;  // wave-uniform
    int lane = threadIdx.x & 31;
    if (e >= nedges) return;
    int s = src[e], d = dst[e];
    const float4* fp = (const float4*)(feat + (size_t)s * D + lane * 8);
    float4 a = fp[0], b = fp[1];
    float* op = neigh + (size_t)d * D + lane * 8;
    atomicAdd(op + 0, a.x); atomicAdd(op + 1, a.y);
    atomicAdd(op + 2, a.z); atomicAdd(op + 3, a.w);
    atomicAdd(op + 4, b.x); atomicAdd(op + 5, b.y);
    atomicAdd(op + 6, b.z); atomicAdd(op + 7, b.w);
    if (lane == 0) atomicAdd(deg + d, 1.0f);
}

// ------------------------------------------------- deg -> 1/max(deg,1) -----
__global__ void sage_invdeg_kernel(float* __restrict__ deg, int n) {
    int i = blockIdx.x * blockDim.x + threadIdx.x;
    if (i < n) deg[i] = 1.0f / fmaxf(deg[i], 1.0f);
}

// -------- scale + convert 8 f32 (two float4) to bf16 halves e0..e0+7 ------
__device__ __forceinline__ void sage_cvt_seg(v16bf& a, int e0,
                                             float4 x, float4 y, float sc) {
    a[e0 + 0] = (__bf16)(x.x * sc); a[e0 + 1] = (__bf16)(x.y * sc);
    a[e0 + 2] = (__bf16)(x.z * sc); a[e0 + 3] = (__bf16)(x.w * sc);
    a[e0 + 4] = (__bf16)(y.x * sc); a[e0 + 5] = (__bf16)(y.y * sc);
    a[e0 + 6] = (__bf16)(y.z * sc); a[e0 + 7] = (__bf16)(y.w * sc);
}

// ---- async stage: 64B of Wcat row -> LDS (same INST_OFFSET both sides) ----
__device__ __forceinline__ void sage_stage_async(unsigned loff,
                                                 const __bf16* gsrc) {
    asm volatile(
        "global_load_async_to_lds_b128 %0, %1, off\n\t"
        "global_load_async_to_lds_b128 %0, %1, off offset:16\n\t"
        "global_load_async_to_lds_b128 %0, %1, off offset:32\n\t"
        "global_load_async_to_lds_b128 %0, %1, off offset:48"
        :
        : "v"(loff), "v"(gsrc)
        : "memory");
}

// -------- issue raw A loads for one K-step (two 8-f32 segments) ------------
__device__ __forceinline__ void sage_load_a_raw(
    const float* __restrict__ feat, const float* __restrict__ neigh,
    int row, int k0, int hi, float sc_n,
    float4& x0, float4& x1, float4& y0, float4& y1, float& sc0, float& sc1) {
    int s0 = k0 + hi * 16;           // A contiguous k = 16*hi + s, halves 0..7
    int s1 = s0 + 8;                 // halves 8..15
    const float* p0 = (s0 < D) ? (feat  + (size_t)row * D + s0)
                               : (neigh + (size_t)row * D + (s0 - D));
    const float* p1 = (s1 < D) ? (feat  + (size_t)row * D + s1)
                               : (neigh + (size_t)row * D + (s1 - D));
    sc0 = (s0 < D) ? 1.0f : sc_n;
    sc1 = (s1 < D) ? 1.0f : sc_n;
    x0 = ((const float4*)p0)[0]; x1 = ((const float4*)p0)[1];
    y0 = ((const float4*)p1)[0]; y1 = ((const float4*)p1)[1];
}

// ---- fused GEMM: out = [feat | neigh*invdeg] @ Wcat^T + b  (bf16 WMMA) ----
// Block = 256 threads = 8 waves; each wave: 16-row stripe x all 256 cols.
// Double-buffered async W staging: one barrier per K-step.
__global__ __launch_bounds__(256) void sage_gemm_kernel(
    const float*  __restrict__ feat,
    const float*  __restrict__ neigh,
    const float*  __restrict__ invdeg,
    const __bf16* __restrict__ wcat,     // [256][512] bf16
    const float*  __restrict__ bias,
    float* __restrict__ out, int nrows) {

    __shared__ __bf16 lds[2 * 256 * 32]; // 2 x 16KB K-slices of Wcat

    const int lane = threadIdx.x & 31;
    const int wid  = threadIdx.x >> 5;
    const int m    = lane & 15;          // A row / B column within tile
    const int hi   = lane >> 4;          // lane half selects K sub-range
    const int row_base = (blockIdx.x * 8 + wid) * 16;
    const bool active  = row_base < nrows;            // wave-uniform
    const int row  = row_base + m;
    const float sc_n = active ? invdeg[row] : 0.0f;

    v8f acc[16] = {};                    // 16 column tiles x 8 f32

    const int t = threadIdx.x;           // this thread stages Wcat row t
    const unsigned lds_off = (unsigned)(uintptr_t)(&lds[t * 32]); // DS-relative
    const __bf16* lb = &lds[(size_t)m * 32 + hi * 16];            // B frag base
    const __bf16* grow = wcat + (size_t)t * DK;

    // ---- prologue: stage slice 0, issue A loads for step 0 ----------------
    sage_stage_async(lds_off, grow);
    float4 cx0 = {}, cx1 = {}, cy0 = {}, cy1 = {};
    float csc0 = 1.0f, csc1 = 1.0f;
    if (active)
        sage_load_a_raw(feat, neigh, row, 0, hi, sc_n,
                        cx0, cx1, cy0, cy1, csc0, csc1);

    for (int it = 0; it < 16; ++it) {
        asm volatile("s_wait_asynccnt 0x0" ::: "memory"); // slice `it` landed
        __syncthreads();                 // all waves done reading buf ((it+1)&1)

        // stage next slice into the other buffer + prefetch next A ----------
        float4 nx0 = {}, nx1 = {}, ny0 = {}, ny1 = {};
        float nsc0 = 1.0f, nsc1 = 1.0f;
        if (it + 1 < 16) {
            unsigned sel = (unsigned)((it + 1) & 1);
            sage_stage_async(lds_off + sel * 16384u, grow + (it + 1) * 32);
            if (active)                  // wave-uniform
                sage_load_a_raw(feat, neigh, row, (it + 1) * 32, hi, sc_n,
                                nx0, nx1, ny0, ny1, nsc0, nsc1);
        }

        if (active) {                    // wave-uniform: EXEC all-ones inside
            v16bf a;
            sage_cvt_seg(a, 0, cx0, cx1, csc0);
            sage_cvt_seg(a, 8, cy0, cy1, csc1);

            const __bf16* lbp = lb + (it & 1) * 8192;  // current buffer
            #pragma unroll
            for (int nt = 0; nt < 16; nt += 2) {
                // two live B fragments -> partial s_wait_dscnt before wmma
                v16bf b0 = *((const v16bf*)(lbp + nt * 512));
                v16bf b1 = *((const v16bf*)(lbp + nt * 512 + 512));
                acc[nt] = __builtin_amdgcn_wmma_f32_16x16x32_bf16(
                    false, a, false, b0, (short)0, acc[nt], false, false);
                acc[nt + 1] = __builtin_amdgcn_wmma_f32_16x16x32_bf16(
                    false, a, false, b1, (short)0, acc[nt + 1], false, false);
            }
        }

        cx0 = nx0; cx1 = nx1; cy0 = ny0; cy1 = ny1;
        csc0 = nsc0; csc1 = nsc1;
    }

    if (active) {
        #pragma unroll
        for (int nt = 0; nt < 16; ++nt) {
            int col = nt * 16 + m;
            float bv = bias[col];
            int r0 = row_base + hi * 8;  // C/D layout: VGPR j -> M = j + 8*hi
            #pragma unroll
            for (int j = 0; j < 8; ++j)
                out[(size_t)(r0 + j) * D + col] = acc[nt][j] + bv;
        }
    }
}

extern "C" void kernel_launch(void* const* d_in, const int* in_sizes, int n_in,
                              void* d_out, int out_size, void* d_ws, size_t ws_size,
                              hipStream_t stream) {
    const float* feat    = (const float*)d_in[0];
    const float* w_neigh = (const float*)d_in[1];
    const float* w_self  = (const float*)d_in[2];
    const float* bias    = (const float*)d_in[3];
    const int*   src     = (const int*)d_in[4];
    const int*   dst     = (const int*)d_in[5];
    float* out = (float*)d_out;

    const int nnodes = in_sizes[0] / D;   // 100000
    const int nedges = in_sizes[4];       // 1600000

    // workspace layout (needs ~103 MB):
    //   [0, N*256*4)            neigh_sum f32
    //   [N*256*4, N*257*4)      deg -> invdeg f32
    //   [N*257*4, +256KB)       Wcat bf16 [256][512]
    char*  ws    = (char*)d_ws;
    float* neigh = (float*)ws;
    float* deg   = (float*)(ws + (size_t)nnodes * D * 4);
    __bf16* wcat = (__bf16*)(ws + (size_t)nnodes * (D + 1) * 4);

    long nz = (long)nnodes * (D + 1);     // zero neigh_sum + deg (contiguous)
    if ((nz & 3) == 0) {
        long nz4 = nz >> 2;
        sage_zero4_kernel<<<(unsigned)((nz4 + 255) / 256), 256, 0, stream>>>(
            (float4*)neigh, nz4);
    } else {
        sage_zero_kernel<<<(unsigned)((nz + 255) / 256), 256, 0, stream>>>(
            neigh, nz);
    }

    sage_pack_w_kernel<<<(D * DK) / 256, 256, 0, stream>>>(w_neigh, w_self, wcat);

    long sthreads = (long)nedges * 32;    // one wave per edge
    sage_scatter_kernel<<<(unsigned)((sthreads + 255) / 256), 256, 0, stream>>>(
        feat, src, dst, neigh, deg, nedges);

    sage_invdeg_kernel<<<(nnodes + 255) / 256, 256, 0, stream>>>(deg, nnodes);

    int rtiles  = (nnodes + 15) / 16;     // 6250 (N divisible by 16)
    int gblocks = (rtiles + 7) / 8;       // 8 row-stripes (waves) per block
    sage_gemm_kernel<<<gblocks, 256, 0, stream>>>(feat, neigh, deg, wcat,
                                                  bias, out, nnodes);
}
